// TimeVAEDecoder_23673859736212
// MI455X (gfx1250) — matrix operand order repaired and checked by reading, not compiled, not run on hardware
//
#include <hip/hip_runtime.h>

typedef __attribute__((ext_vector_type(16))) __bf16 v16bf;
typedef __attribute__((ext_vector_type(8)))  float  v8f;

namespace {
constexpr int cL    = 128;
constexpr int cH    = 1024;
constexpr int cO    = 64;
constexpr int cS    = 256;
constexpr int cB    = 512;
constexpr int cIN0  = cL + cO;        // 192 = z | fed-back out
constexpr int cKT0A = cIN0 / 32;      // 6 k-tiles over X0 (z|out)
constexpr int cKT0B = cH / 32;        // 32 k-tiles over h1
constexpr int cKT0  = cKT0A + cKT0B;  // 38
constexpr int cKT1  = (2 * cH) / 32;  // 64
constexpr int cKTO  = cH / 32;        // 32
constexpr int cNG   = 4 * cH;         // 4096 gate columns
constexpr int cNTG  = cNG / 16;       // 256 n-tiles
constexpr int cNTO  = cO / 16;        // 4
constexpr int cX1LD = 4 * cH;         // h1 p0 | h1 p1 | h2 p0 | h2 p1
constexpr int NWG   = 128;            // 8 row-blocks(64) x 16 j-slices(64)
}

// ---------------- device helpers ----------------

__device__ __forceinline__ unsigned short f2bf(float f) {
  unsigned u = __float_as_uint(f);
  unsigned r = ((u >> 16) & 1u) + 0x7fffu;   // round-to-nearest-even
  return (unsigned short)((u + r) >> 16);
}
__device__ __forceinline__ float sigm(float x) { return 1.f / (1.f + __expf(-x)); }
__device__ __forceinline__ float ftanh(float x) {
  x = fminf(fmaxf(x, -15.f), 15.f);
  float e = __expf(2.f * x);
  return (e - 1.f) / (e + 1.f);
}

// A fragment (16x32 bf16) from a per-lane base pointer (two 16B loads, +0/+16B).
__device__ __forceinline__ v16bf load_a_p(const unsigned short* __restrict__ p) {
  union { uint4 u[2]; v16bf v; } t;
  t.u[0] = *(const uint4*)(p);
  t.u[1] = *(const uint4*)(p + 8);
  return t.v;
}

// B fragment from a packed blob per-lane pointer (32 contiguous bytes).
__device__ __forceinline__ v16bf load_b_p(const unsigned short* __restrict__ p) {
  const uint4* q = (const uint4*)p;
  union { uint4 u[2]; v16bf v; } t;
  t.u[0] = q[0];
  t.u[1] = q[1];
  return t.v;
}

__device__ __forceinline__ v8f wmma_bf16(v16bf a, v16bf b, v8f c) {
  return __builtin_amdgcn_wmma_f32_16x16x32_bf16(false, a, false, b, (short)0, c,
                                                 false, false);
}

// Device-scope sense-reversing grid barrier.
__device__ __forceinline__ void grid_barrier(unsigned* __restrict__ bar, unsigned nwg) {
  __syncthreads();
  if (threadIdx.x == 0) {
    __threadfence();
    unsigned g = __hip_atomic_load(&bar[1], __ATOMIC_RELAXED, __HIP_MEMORY_SCOPE_AGENT);
    unsigned a = __hip_atomic_fetch_add(&bar[0], 1u, __ATOMIC_ACQ_REL,
                                        __HIP_MEMORY_SCOPE_AGENT);
    if (a == nwg - 1u) {
      __hip_atomic_store(&bar[0], 0u, __ATOMIC_RELAXED, __HIP_MEMORY_SCOPE_AGENT);
      __hip_atomic_fetch_add(&bar[1], 1u, __ATOMIC_RELEASE, __HIP_MEMORY_SCOPE_AGENT);
    } else {
      while (__hip_atomic_load(&bar[1], __ATOMIC_ACQUIRE,
                               __HIP_MEMORY_SCOPE_AGENT) == g) {
        __builtin_amdgcn_s_sleep(2);
      }
    }
    __threadfence();
  }
  __syncthreads();
}

// ---------------- prep kernels ----------------

__global__ void pack_w_kernel(const float* __restrict__ Wa, const float* __restrict__ Wb,
                              int Ka, int Kb, int ktiles, int ntiles,
                              unsigned short* __restrict__ dst) {
  int tid = blockIdx.x * blockDim.x + threadIdx.x;
  int total = ktiles * ntiles * 512;
  if (tid >= total) return;
  int blob = tid >> 9;
  int pos  = tid & 511;
  int l = pos >> 4, t = pos & 15;
  int kt = blob / ntiles, nt = blob % ntiles;
  int k = kt * 32 + ((l >> 4) << 4) + t;   // lanes>=16 carry K+16
  int n = nt * 16 + (l & 15);
  float v = (k < Ka) ? Wa[(size_t)n * Ka + k] : Wb[(size_t)n * Kb + (k - Ka)];
  dst[tid] = f2bf(v);
}

__global__ void init_misc_kernel(const float* __restrict__ bih0, const float* __restrict__ bhh0,
                                 const float* __restrict__ bih1, const float* __restrict__ bhh1,
                                 float* __restrict__ B0, float* __restrict__ B1,
                                 unsigned* __restrict__ bar) {
  int tid = blockIdx.x * blockDim.x + threadIdx.x;
  if (tid < cNG) {
    B0[tid] = bih0[tid] + bhh0[tid];
    B1[tid] = bih1[tid] + bhh1[tid];
  }
  if (tid < 2) bar[tid] = 0u;
}

__global__ void init_x0_kernel(const float* __restrict__ z, unsigned short* __restrict__ X0) {
  int tid = blockIdx.x * blockDim.x + threadIdx.x;
  if (tid >= cB * cIN0) return;
  int row = tid / cIN0, col = tid % cIN0;
  X0[tid] = (col < cL) ? f2bf(z[(size_t)row * cL + col]) : (unsigned short)0;
}

__global__ void init_state_kernel(const float* __restrict__ z,
                                  const float* __restrict__ Wh, const float* __restrict__ bh,
                                  const float* __restrict__ Wc, const float* __restrict__ bc,
                                  unsigned short* __restrict__ X1, float* __restrict__ CI) {
  int tid = blockIdx.x * blockDim.x + threadIdx.x;
  if (tid >= cB * cH) return;
  int row = tid / cH, j = tid % cH;
  const float* zr = z + (size_t)row * cL;
  const float* wh = Wh + (size_t)j * cL;
  const float* wc = Wc + (size_t)j * cL;
  float h = bh[j], c = bc[j];
  for (int k = 0; k < cL; ++k) {
    h = fmaf(zr[k], wh[k], h);
    c = fmaf(zr[k], wc[k], c);
  }
  unsigned short hb = f2bf(h);
  X1[(size_t)row * cX1LD + j] = hb;            // h1, plane 0
  X1[(size_t)row * cX1LD + 2 * cH + j] = hb;   // h2, plane 0
  CI[tid] = c;
}

// ---------------- persistent recurrence kernel ----------------
// Grid: 128 WGs x 256 threads (8 wave32). WG = 64 batch rows x 64 hidden units.
// Wave (wr=w>>2, wc=w&3): 32 rows (two 16-row M-tiles) x 16 units; each B
// fragment loaded once feeds 2 WMMAs -> halves L2 weight traffic per step.
__global__ __launch_bounds__(256, 1) void lstm_persistent_kernel(
    const unsigned short* __restrict__ W0P,
    const unsigned short* __restrict__ W1P,
    const unsigned short* __restrict__ WOP,
    unsigned short* __restrict__ X0,
    unsigned short* __restrict__ X1,
    const float* __restrict__ CI,
    const float* __restrict__ B0,
    const float* __restrict__ B1,
    const float* __restrict__ bout,
    float* __restrict__ out,
    unsigned* __restrict__ bar) {
  __shared__ float red[8 * 256];
  const int tid  = threadIdx.x;
  const int lane = tid & 31;
  const int w    = tid >> 5;
  const int wr   = w >> 2;               // 0..1 (32-row half)
  const int wc   = w & 3;                // 0..3 (16-unit slice)
  const int bid  = blockIdx.x;
  const int wgM  = bid >> 4;             // 0..7  (64-row blocks)
  const int wgJ  = bid & 15;             // 0..15 (64-unit slices)
  const int mrow0 = wgM * 64 + wr * 32;  // 32-row base for this wave (2 M-tiles)
  const int jbase = wgJ * 64 + wc * 16;  // hidden-unit base for this wave
  const int mtO = bid >> 2, ntO = bid & 3;  // out-proj tile
  const int rloc = (lane >> 4) << 3;     // C/D layout: lanes>=16 -> M+8
  const int ncol = lane & 15;

  // Per-lane A-fragment addressing constants (two M-tiles: +0 and +16 rows).
  const int arow = mrow0 + (lane & 15);
  const int aoff = (lane >> 4) << 4;     // 0 or 16 (K element offset)

  // Per-lane packed-B base for this wave's j-slice (first quadrant).
  const unsigned short* bbase0 = W0P + ((size_t)(jbase >> 4) << 9) + (lane << 4);
  const unsigned short* bbase1 = W1P + ((size_t)(jbase >> 4) << 9) + (lane << 4);
  constexpr size_t QSTRIDE = (size_t)64 << 9;     // quadrant n-tile stride
  constexpr size_t KSTRIDE = (size_t)cNTG << 9;   // k-tile stride

  // Cell states live in registers for the whole sequence. [m-tile][r]
  float c1[2][8], c2[2][8];
#pragma unroll
  for (int m = 0; m < 2; ++m)
#pragma unroll
    for (int r = 0; r < 8; ++r) {
      float cv = CI[(size_t)(mrow0 + m * 16 + rloc + r) * cH + jbase + ncol];
      c1[m][r] = cv;
      c2[m][r] = cv;
    }

  for (int s = 0; s < cS; ++s) {
    const int rp = s & 1;  // read plane; write plane = 1-rp

    // ---- phase A: layer-0 gates + pointwise ----
    {
      v8f a00 = v8f{}, a01 = v8f{}, a02 = v8f{}, a03 = v8f{};  // m-tile 0, q0..3
      v8f a10 = v8f{}, a11 = v8f{}, a12 = v8f{}, a13 = v8f{};  // m-tile 1, q0..3
      // segment 1: K over X0 = [z | out], 6 k-tiles
      {
        const unsigned short* ap0 = X0 + (size_t)arow * cIN0 + aoff;
        const unsigned short* ap1 = ap0 + (size_t)16 * cIN0;
        const unsigned short* bp  = bbase0;
        for (int kt = 0; kt < cKT0A; ++kt) {
          v16bf a0 = load_a_p(ap0);
          v16bf a1 = load_a_p(ap1);
          v16bf b0 = load_b_p(bp);
          v16bf b1 = load_b_p(bp + QSTRIDE);
          v16bf b2 = load_b_p(bp + 2 * QSTRIDE);
          v16bf b3 = load_b_p(bp + 3 * QSTRIDE);
          a00 = wmma_bf16(a0, b0, a00); a10 = wmma_bf16(a1, b0, a10);
          a01 = wmma_bf16(a0, b1, a01); a11 = wmma_bf16(a1, b1, a11);
          a02 = wmma_bf16(a0, b2, a02); a12 = wmma_bf16(a1, b2, a12);
          a03 = wmma_bf16(a0, b3, a03); a13 = wmma_bf16(a1, b3, a13);
          ap0 += 32; ap1 += 32; bp += KSTRIDE;
        }
      }
      // segment 2: K over h1 (plane rp), 32 k-tiles
      {
        const unsigned short* ap0 = X1 + (size_t)arow * cX1LD + rp * cH + aoff;
        const unsigned short* ap1 = ap0 + (size_t)16 * cX1LD;
        const unsigned short* bp  = bbase0 + (size_t)cKT0A * KSTRIDE;
        for (int kt = 0; kt < cKT0B; ++kt) {
          v16bf a0 = load_a_p(ap0);
          v16bf a1 = load_a_p(ap1);
          v16bf b0 = load_b_p(bp);
          v16bf b1 = load_b_p(bp + QSTRIDE);
          v16bf b2 = load_b_p(bp + 2 * QSTRIDE);
          v16bf b3 = load_b_p(bp + 3 * QSTRIDE);
          a00 = wmma_bf16(a0, b0, a00); a10 = wmma_bf16(a1, b0, a10);
          a01 = wmma_bf16(a0, b1, a01); a11 = wmma_bf16(a1, b1, a11);
          a02 = wmma_bf16(a0, b2, a02); a12 = wmma_bf16(a1, b2, a12);
          a03 = wmma_bf16(a0, b3, a03); a13 = wmma_bf16(a1, b3, a13);
          ap0 += 32; ap1 += 32; bp += KSTRIDE;
        }
      }
      // pointwise LSTM (register-local i/f/g/o)
      const int j = jbase + ncol;
      const float bi = B0[j], bff = B0[cH + j], bg = B0[2 * cH + j], bo = B0[3 * cH + j];
      unsigned short* hdst = X1 + (size_t)(mrow0 + rloc) * cX1LD + (1 - rp) * cH + j;
#pragma unroll
      for (int r = 0; r < 8; ++r) {
        float cn = sigm(a01[r] + bff) * c1[0][r] + sigm(a00[r] + bi) * ftanh(a02[r] + bg);
        float hn = sigm(a03[r] + bo) * ftanh(cn);
        c1[0][r] = cn;
        hdst[(size_t)r * cX1LD] = f2bf(hn);
      }
      hdst += (size_t)16 * cX1LD;
#pragma unroll
      for (int r = 0; r < 8; ++r) {
        float cn = sigm(a11[r] + bff) * c1[1][r] + sigm(a10[r] + bi) * ftanh(a12[r] + bg);
        float hn = sigm(a13[r] + bo) * ftanh(cn);
        c1[1][r] = cn;
        hdst[(size_t)r * cX1LD] = f2bf(hn);
      }
    }
    grid_barrier(bar, NWG);

    // ---- phase B: layer-1 gates + pointwise ----
    {
      v8f a00 = v8f{}, a01 = v8f{}, a02 = v8f{}, a03 = v8f{};
      v8f a10 = v8f{}, a11 = v8f{}, a12 = v8f{}, a13 = v8f{};
      // segment 1: K over h1 new plane, 32 k-tiles
      {
        const unsigned short* ap0 = X1 + (size_t)arow * cX1LD + (1 - rp) * cH + aoff;
        const unsigned short* ap1 = ap0 + (size_t)16 * cX1LD;
        const unsigned short* bp  = bbase1;
        for (int kt = 0; kt < 32; ++kt) {
          v16bf a0 = load_a_p(ap0);
          v16bf a1 = load_a_p(ap1);
          v16bf b0 = load_b_p(bp);
          v16bf b1 = load_b_p(bp + QSTRIDE);
          v16bf b2 = load_b_p(bp + 2 * QSTRIDE);
          v16bf b3 = load_b_p(bp + 3 * QSTRIDE);
          a00 = wmma_bf16(a0, b0, a00); a10 = wmma_bf16(a1, b0, a10);
          a01 = wmma_bf16(a0, b1, a01); a11 = wmma_bf16(a1, b1, a11);
          a02 = wmma_bf16(a0, b2, a02); a12 = wmma_bf16(a1, b2, a12);
          a03 = wmma_bf16(a0, b3, a03); a13 = wmma_bf16(a1, b3, a13);
          ap0 += 32; ap1 += 32; bp += KSTRIDE;
        }
      }
      // segment 2: K over h2 old plane, 32 k-tiles
      {
        const unsigned short* ap0 = X1 + (size_t)arow * cX1LD + 2 * cH + rp * cH + aoff;
        const unsigned short* ap1 = ap0 + (size_t)16 * cX1LD;
        const unsigned short* bp  = bbase1 + (size_t)32 * KSTRIDE;
        for (int kt = 0; kt < 32; ++kt) {
          v16bf a0 = load_a_p(ap0);
          v16bf a1 = load_a_p(ap1);
          v16bf b0 = load_b_p(bp);
          v16bf b1 = load_b_p(bp + QSTRIDE);
          v16bf b2 = load_b_p(bp + 2 * QSTRIDE);
          v16bf b3 = load_b_p(bp + 3 * QSTRIDE);
          a00 = wmma_bf16(a0, b0, a00); a10 = wmma_bf16(a1, b0, a10);
          a01 = wmma_bf16(a0, b1, a01); a11 = wmma_bf16(a1, b1, a11);
          a02 = wmma_bf16(a0, b2, a02); a12 = wmma_bf16(a1, b2, a12);
          a03 = wmma_bf16(a0, b3, a03); a13 = wmma_bf16(a1, b3, a13);
          ap0 += 32; ap1 += 32; bp += KSTRIDE;
        }
      }
      const int j = jbase + ncol;
      const float bi = B1[j], bff = B1[cH + j], bg = B1[2 * cH + j], bo = B1[3 * cH + j];
      unsigned short* hdst =
          X1 + (size_t)(mrow0 + rloc) * cX1LD + 2 * cH + (1 - rp) * cH + j;
#pragma unroll
      for (int r = 0; r < 8; ++r) {
        float cn = sigm(a01[r] + bff) * c2[0][r] + sigm(a00[r] + bi) * ftanh(a02[r] + bg);
        float hn = sigm(a03[r] + bo) * ftanh(cn);
        c2[0][r] = cn;
        hdst[(size_t)r * cX1LD] = f2bf(hn);
      }
      hdst += (size_t)16 * cX1LD;
#pragma unroll
      for (int r = 0; r < 8; ++r) {
        float cn = sigm(a11[r] + bff) * c2[1][r] + sigm(a10[r] + bi) * ftanh(a12[r] + bg);
        float hn = sigm(a13[r] + bo) * ftanh(cn);
        c2[1][r] = cn;
        hdst[(size_t)r * cX1LD] = f2bf(hn);
      }
    }
    grid_barrier(bar, NWG);

    // ---- phase C: out = h2_new Wout^T + bout ; feed back into X0 ----
    {
      v8f acc = v8f{};
      {
        const unsigned short* ap =
            X1 + (size_t)(mtO * 16 + (lane & 15)) * cX1LD + 2 * cH + (1 - rp) * cH +
            (size_t)w * 4 * 32 + aoff;
        const unsigned short* bp =
            WOP + ((size_t)(w * 4 * cNTO + ntO) << 9) + (lane << 4);
        for (int kt = 0; kt < 4; ++kt) {  // K split across the 8 waves
          v16bf a = load_a_p(ap);
          acc = wmma_bf16(a, load_b_p(bp), acc);
          ap += 32;
          bp += (size_t)cNTO << 9;
        }
      }
#pragma unroll
      for (int r = 0; r < 8; ++r) red[w * 256 + r * 32 + lane] = acc[r];
      __syncthreads();
      {
        float sum = 0.f;
#pragma unroll
        for (int ww = 0; ww < 8; ++ww) sum += red[ww * 256 + tid];
        int lr = tid & 31, rr = tid >> 5;
        int m = rr + ((lr >> 4) << 3);
        int n = lr & 15;
        float val = sum + bout[ntO * 16 + n];
        int row = mtO * 16 + m;
        out[(size_t)row * (cS * cO) + (size_t)s * cO + ntO * 16 + n] = val;
        X0[(size_t)row * cIN0 + cL + ntO * 16 + n] = f2bf(val);
      }
    }
    grid_barrier(bar, NWG);
  }
}

// ---------------- host ----------------

extern "C" void kernel_launch(void* const* d_in, const int* in_sizes, int n_in,
                              void* d_out, int out_size, void* d_ws, size_t ws_size,
                              hipStream_t stream) {
  (void)in_sizes; (void)n_in; (void)out_size; (void)ws_size;
  const float* z    = (const float*)d_in[0];
  const float* Wh   = (const float*)d_in[1];
  const float* bh   = (const float*)d_in[2];
  const float* Wc   = (const float*)d_in[3];
  const float* bc   = (const float*)d_in[4];
  const float* Wih0 = (const float*)d_in[5];
  const float* Whh0 = (const float*)d_in[6];
  const float* bih0 = (const float*)d_in[7];
  const float* bhh0 = (const float*)d_in[8];
  const float* Wih1 = (const float*)d_in[9];
  const float* Whh1 = (const float*)d_in[10];
  const float* bih1 = (const float*)d_in[11];
  const float* bhh1 = (const float*)d_in[12];
  const float* Wout = (const float*)d_in[13];
  const float* bout = (const float*)d_in[14];
  float* outp = (float*)d_out;

  unsigned char* base = (unsigned char*)d_ws;
  size_t off = 0;
  auto take = [&](size_t n) {
    size_t o = off;
    off += (n + 1023) & ~(size_t)1023;
    return o;
  };
  unsigned*       bar = (unsigned*)(base + take(256));
  unsigned short* W0P = (unsigned short*)(base + take((size_t)cKT0 * cNTG * 1024));
  unsigned short* W1P = (unsigned short*)(base + take((size_t)cKT1 * cNTG * 1024));
  unsigned short* WOP = (unsigned short*)(base + take((size_t)cKTO * cNTO * 1024));
  unsigned short* X0  = (unsigned short*)(base + take((size_t)cB * cIN0 * 2));
  unsigned short* X1  = (unsigned short*)(base + take((size_t)cB * cX1LD * 2));
  float*          CIp = (float*)(base + take((size_t)cB * cH * 4));
  float*          B0p = (float*)(base + take((size_t)cNG * 4));
  float*          B1p = (float*)(base + take((size_t)cNG * 4));

  pack_w_kernel<<<(cKT0 * cNTG * 512) / 256, 256, 0, stream>>>(
      Wih0, Whh0, cIN0, cH, cKT0, cNTG, W0P);
  pack_w_kernel<<<(cKT1 * cNTG * 512) / 256, 256, 0, stream>>>(
      Wih1, Whh1, cH, cH, cKT1, cNTG, W1P);
  pack_w_kernel<<<(cKTO * cNTO * 512) / 256, 256, 0, stream>>>(
      Wout, Wout, cH, cH, cKTO, cNTO, WOP);
  init_misc_kernel<<<(cNG + 255) / 256, 256, 0, stream>>>(
      bih0, bhh0, bih1, bhh1, B0p, B1p, bar);
  init_x0_kernel<<<(cB * cIN0) / 256, 256, 0, stream>>>(z, X0);
  init_state_kernel<<<(cB * cH) / 256, 256, 0, stream>>>(z, Wh, bh, Wc, bc, X1, CIp);

  lstm_persistent_kernel<<<NWG, 256, 0, stream>>>(
      W0P, W1P, WOP, X0, X1, CIp, B0p, B1p, bout, outp, bar);
}
